// SetAbstraction_89936615178985
// MI455X (gfx1250) — compile-verified, matrix operand order checked by reading, not run on hardware
//
#include <hip/hip_runtime.h>
#include <math.h>

typedef __attribute__((ext_vector_type(16))) __bf16 v16bf;
typedef __attribute__((ext_vector_type(8)))  float  v8f;

#define NB 8
#define NP 16384
#define NC 64
#define NS 2048
#define NK 32
#define HSTRIDE 136   // padded LDS row stride in bf16 units (128 + 8) -> 272B rows, 16B aligned

// ---------------------------------------------------------------------------
// Kernel 1: convert weights to bf16, transposed [dout][din], zero-pad W1 rows
// 124..127 so the GEMM can run with K padded to 128.
// ---------------------------------------------------------------------------
__global__ void prep_weights_kernel(const float* __restrict__ W1,
                                    const float* __restrict__ W2,
                                    const float* __restrict__ W3,
                                    __bf16* __restrict__ Wt1,
                                    __bf16* __restrict__ Wt2,
                                    __bf16* __restrict__ Wt3) {
  int t = blockIdx.x * blockDim.x + threadIdx.x;
  if (t < 128 * 128) {
    int o = t >> 7, i = t & 127;
    float v = (i < 124) ? W1[i * 128 + o] : 0.f;
    Wt1[o * 128 + i] = (__bf16)v;
    Wt2[o * 128 + i] = (__bf16)W2[i * 128 + o];
  }
  if (t < 256 * 128) {
    int o = t >> 7, i = t & 127;
    Wt3[o * 128 + i] = (__bf16)W3[i * 256 + o];
  }
}

// ---------------------------------------------------------------------------
// Kernel 2: farthest point sampling. One 1024-thread block per batch; each
// thread keeps 16 points (xyz + running min-dist) in registers; block-wide
// argmax through LDS each of the 2048 sequential steps. Writes new_xyz.
// ---------------------------------------------------------------------------
__global__ __launch_bounds__(1024) void fps_kernel(const float* __restrict__ xyz,
                                                   float* __restrict__ out_xyz) {
  const int b = blockIdx.x;
  const int t = threadIdx.x;
  const float* bx = xyz + (size_t)b * NP * 3;

  float px[16], py[16], pz[16], pd[16];
#pragma unroll
  for (int i = 0; i < 16; ++i) {
    int p = t + i * 1024;
    px[i] = bx[p * 3 + 0];
    py[i] = bx[p * 3 + 1];
    pz[i] = bx[p * 3 + 2];
    pd[i] = 1e10f;
  }

  __shared__ float sd[1024];
  __shared__ int   si[1024];
  __shared__ float scent[3];

  int far = 0;
  for (int s = 0; s < NS; ++s) {
    if (t == 0) {
      float cx = bx[far * 3 + 0], cy = bx[far * 3 + 1], cz = bx[far * 3 + 2];
      scent[0] = cx; scent[1] = cy; scent[2] = cz;
      float* o = out_xyz + ((size_t)b * NS + s) * 3;
      o[0] = cx; o[1] = cy; o[2] = cz;
    }
    __syncthreads();
    const float cx = scent[0], cy = scent[1], cz = scent[2];
    float lmax = -1.f; int lidx = t;
#pragma unroll
    for (int i = 0; i < 16; ++i) {
      float dx = px[i] - cx, dy = py[i] - cy, dz = pz[i] - cz;
      float d = dx * dx + dy * dy + dz * dz;
      pd[i] = fminf(pd[i], d);
      if (pd[i] > lmax) { lmax = pd[i]; lidx = t + i * 1024; }
    }
    sd[t] = lmax; si[t] = lidx;
    __syncthreads();
    for (int off = 512; off > 0; off >>= 1) {
      if (t < off) {
        float od = sd[t + off]; int oi = si[t + off];
        if (od > sd[t] || (od == sd[t] && oi < si[t])) { sd[t] = od; si[t] = oi; }
      }
      __syncthreads();
    }
    far = si[0];
    __syncthreads();
  }
}

// ---------------------------------------------------------------------------
// Kernel 3: KNN (unordered top-32 nearest). One wave32 per centroid; each
// lane owns one slot of the 32-entry pool. Candidates that beat the pool max
// are inserted one at a time (ballot-serialized), pool max recomputed with a
// lane-consistent shuffle argmax. Points 512 ahead (16 chunks) prefetched into cache.
// ---------------------------------------------------------------------------
__global__ __launch_bounds__(256) void knn_kernel(const float* __restrict__ xyz,
                                                  const float* __restrict__ new_xyz,
                                                  int* __restrict__ knn) {
  const int gw = (blockIdx.x * 256 + threadIdx.x) >> 5;   // centroid id
  const int lane = threadIdx.x & 31;
  if (gw >= NB * NS) return;
  const int b = gw / NS;
  const float* bx = xyz + (size_t)b * NP * 3;
  const float* c = new_xyz + (size_t)gw * 3;
  const float cx = c[0], cy = c[1], cz = c[2];

  // pool init with first 32 points
  int pi = lane;
  float dx = bx[lane * 3 + 0] - cx, dy = bx[lane * 3 + 1] - cy, dz = bx[lane * 3 + 2] - cz;
  float pdv = dx * dx + dy * dy + dz * dz;

  float md = pdv; int ml = lane;
#pragma unroll
  for (int off = 16; off > 0; off >>= 1) {
    float od = __shfl_xor(md, off);
    int   ol = __shfl_xor(ml, off);
    if (od > md || (od == md && ol > ml)) { md = od; ml = ol; }
  }

  for (int j = 32; j < NP; j += 32) {
    const int p = j + lane;
    if (j + 512 < NP) __builtin_prefetch(&bx[(p + 512) * 3], 0, 0);
    float ax = bx[p * 3 + 0] - cx, ay = bx[p * 3 + 1] - cy, az = bx[p * 3 + 2] - cz;
    float d = ax * ax + ay * ay + az * az;
    unsigned long long m = __ballot(d < md);
    while (m) {
      int f = __ffsll((unsigned long long)m) - 1;
      m &= m - 1;
      float fd = __shfl(d, f);
      if (fd < md) {
        if (lane == ml) { pdv = fd; pi = j + f; }
        md = pdv; ml = lane;
#pragma unroll
        for (int off = 16; off > 0; off >>= 1) {
          float od = __shfl_xor(md, off);
          int   ol = __shfl_xor(ml, off);
          if (od > md || (od == md && ol > ml)) { md = od; ml = ol; }
        }
      }
    }
  }
  knn[(size_t)gw * NK + lane] = pi;
}

// ---------------------------------------------------------------------------
// WMMA fragment loaders (bf16, 16x16x32). LDS row-major with HSTRIDE pad.
// A 16x32 layout: lane half h holds K-blocks {8h..8h+7, 16+8h..16+8h+7};
// pairs of consecutive K per dword => one 32-bit LDS read per dword.
// B 32x16 uses the mirrored layout (lanes -> N, same K pattern) against
// weights stored [dout][din].
// ---------------------------------------------------------------------------
__device__ __forceinline__ v16bf load_frag(const __bf16* base, int row0, int kbase, int lane) {
  const int h = (lane >> 4) & 1;
  const int m = lane & 15;
  const __bf16* rp = base + (row0 + m) * HSTRIDE + kbase + 8 * h;
  union { unsigned int u[8]; v16bf v; } r;
#pragma unroll
  for (int j = 0; j < 8; ++j) {
    const int koff = (j < 4) ? (2 * j) : (16 + 2 * (j - 4));
    r.u[j] = *(const unsigned int*)(rp + koff);
  }
  return r.v;
}

// Async DMA weight staging: global -> LDS without touching VGPR data paths.
// Each lane issues B128 copies; row stride 272B keeps LDS side 16B-aligned.
__device__ __forceinline__ void stage_w_async(__bf16* Ws, const __bf16* g, int rows, int t) {
  const int total = rows * 16;              // 16B units per 128-col row
  for (int u = t; u < total; u += 256) {
    const int row = u >> 4, cu = u & 15;
    const unsigned lds = (unsigned)(unsigned long long)(const void*)(Ws + row * HSTRIDE) + cu * 16;
    const void* ga = (const void*)(g + row * 128 + cu * 8);
    asm volatile("global_load_async_to_lds_b128 %0, %1, off"
                 :: "v"(lds), "v"(ga) : "memory");
  }
}

__device__ __forceinline__ void wait_async_all() {
  asm volatile("s_wait_asynccnt 0x0" ::: "memory");
}

// GEMM(128 rows x 128 cols, K=128 bf16) + bias + LayerNorm + ReLU, all in regs.
__device__ __forceinline__ void layer128(const __bf16* Hs, const __bf16* Ws,
                                         const float* Pb, const float* Pg, const float* Pbe,
                                         int wave, int lane, v8f acc[8]) {
#pragma unroll
  for (int ct = 0; ct < 8; ++ct)
#pragma unroll
    for (int r = 0; r < 8; ++r) acc[ct][r] = 0.f;

#pragma unroll
  for (int ks = 0; ks < 4; ++ks) {
    v16bf a = load_frag(Hs, 16 * wave, 32 * ks, lane);
#pragma unroll
    for (int ct = 0; ct < 8; ++ct) {
      v16bf bm = load_frag(Ws, 16 * ct, 32 * ks, lane);
      acc[ct] = __builtin_amdgcn_wmma_f32_16x16x32_bf16(false, a, false, bm,
                                                        (short)0, acc[ct], false, false);
    }
  }
  const int n = lane & 15;
#pragma unroll
  for (int ct = 0; ct < 8; ++ct) {
    const float bias = Pb[ct * 16 + n];
#pragma unroll
    for (int r = 0; r < 8; ++r) acc[ct][r] += bias;
  }
  // Per C/D layout: VGPR r holds one full row striped over 16 lanes of a half,
  // so LN stats are a 16-lane shuffle reduction.
#pragma unroll
  for (int r = 0; r < 8; ++r) {
    float s = 0.f, s2 = 0.f;
#pragma unroll
    for (int ct = 0; ct < 8; ++ct) { float v = acc[ct][r]; s += v; s2 += v * v; }
#pragma unroll
    for (int off = 1; off < 16; off <<= 1) { s += __shfl_xor(s, off); s2 += __shfl_xor(s2, off); }
    const float mean = s * (1.f / 128.f);
    const float var  = s2 * (1.f / 128.f) - mean * mean;
    const float rstd = rsqrtf(var + 1e-5f);
#pragma unroll
    for (int ct = 0; ct < 8; ++ct) {
      const int col = ct * 16 + n;
      float v = (acc[ct][r] - mean) * rstd * Pg[col] + Pbe[col];
      acc[ct][r] = fmaxf(v, 0.f);
    }
  }
}

// ---------------------------------------------------------------------------
// Kernel 4: fused gather + NeRF encode + MLP (3 GEMMs, 2 LN/ReLU) + max-pool.
// Block = 256 threads = 8 waves, owns 4 centroids x 32 samples = 128 rows.
// ---------------------------------------------------------------------------
__global__ __launch_bounds__(256) void sa_mlp_kernel(
    const float* __restrict__ xyz, const float* __restrict__ feat,
    const int* __restrict__ knn, const float* __restrict__ new_xyz,
    const __bf16* __restrict__ Wt1, const __bf16* __restrict__ Wt2,
    const __bf16* __restrict__ Wt3,
    const float* __restrict__ b1, const float* __restrict__ g1, const float* __restrict__ be1,
    const float* __restrict__ b2, const float* __restrict__ g2, const float* __restrict__ be2,
    const float* __restrict__ b3, float* __restrict__ out_feat) {
  __shared__ __bf16 Hs[128 * HSTRIDE];      // activations   (34.8 KB)
  __shared__ __bf16 Ws[256 * HSTRIDE];      // weight stage  (69.6 KB)
  __shared__ float  Pmax[8][256];           // partial maxes ( 8 KB)
  __shared__ float  Pb1[128], Pg1[128], Pbe1[128];
  __shared__ float  Pb2[128], Pg2[128], Pbe2[128];
  __shared__ float  Pb3[256];

  const int t = threadIdx.x;
  const int wave = t >> 5, lane = t & 31;
  const int cent0 = blockIdx.x * 4;

  // kick off async DMA of layer-1 weights first so it overlaps the encode
  stage_w_async(Ws, Wt1, 128, t);

  if (t < 128) {
    Pb1[t] = b1[t]; Pg1[t] = g1[t]; Pbe1[t] = be1[t];
    Pb2[t] = b2[t]; Pg2[t] = g2[t]; Pbe2[t] = be2[t];
  }
  Pb3[t] = b3[t];

  // ---- build H0: cols 0..63 features, 64..123 positional enc, 124..127 zero
  {
    const int r = t >> 1, part = t & 1;
    const int cl = r >> 5, samp = r & 31;
    const int cent = cent0 + cl;
    const int bb = cent >> 11;                     // / NS
    const int j = knn[(size_t)cent * NK + samp];
    const float* pj = xyz + ((size_t)bb * NP + j) * 3;
    const float* cc = new_xyz + (size_t)cent * 3;
    const float rx = pj[0] - cc[0], ry = pj[1] - cc[1], rz = pj[2] - cc[2];
    __bf16* hrow = Hs + r * HSTRIDE;
    if (part == 0) {
      const float4* fp = (const float4*)(feat + ((size_t)bb * NP + j) * NC);
#pragma unroll
      for (int q = 0; q < 16; ++q) {
        float4 v = fp[q];
        hrow[q * 4 + 0] = (__bf16)v.x; hrow[q * 4 + 1] = (__bf16)v.y;
        hrow[q * 4 + 2] = (__bf16)v.z; hrow[q * 4 + 3] = (__bf16)v.w;
      }
    } else {
#pragma unroll
      for (int fb = 0; fb < 10; ++fb) {
        const float fr = (float)(1 << fb);
        float sx, cxx, sy, cyy, sz, czz;
        __sincosf(rx * fr, &sx, &cxx);
        __sincosf(ry * fr, &sy, &cyy);
        __sincosf(rz * fr, &sz, &czz);
        __bf16* e = hrow + 64 + fb * 6;
        e[0] = (__bf16)sx;  e[1] = (__bf16)sy;  e[2] = (__bf16)sz;
        e[3] = (__bf16)cxx; e[4] = (__bf16)cyy; e[5] = (__bf16)czz;
      }
      hrow[124] = (__bf16)0.f; hrow[125] = (__bf16)0.f;
      hrow[126] = (__bf16)0.f; hrow[127] = (__bf16)0.f;
    }
  }
  wait_async_all();
  __syncthreads();

  const int n = lane & 15, h = (lane >> 4) & 1;
  v8f acc[8];

  // ---- layer 1
  layer128(Hs, Ws, Pb1, Pg1, Pbe1, wave, lane, acc);
  __syncthreads();
  stage_w_async(Ws, Wt2, 128, t);
#pragma unroll
  for (int r = 0; r < 8; ++r) {
    const int row = 16 * wave + r + 8 * h;
#pragma unroll
    for (int ct = 0; ct < 8; ++ct) Hs[row * HSTRIDE + ct * 16 + n] = (__bf16)acc[ct][r];
  }
  wait_async_all();
  __syncthreads();

  // ---- layer 2
  layer128(Hs, Ws, Pb2, Pg2, Pbe2, wave, lane, acc);
  __syncthreads();
  stage_w_async(Ws, Wt3, 256, t);
#pragma unroll
  for (int r = 0; r < 8; ++r) {
    const int row = 16 * wave + r + 8 * h;
#pragma unroll
    for (int ct = 0; ct < 8; ++ct) Hs[row * HSTRIDE + ct * 16 + n] = (__bf16)acc[ct][r];
  }
  wait_async_all();
  __syncthreads();

  // ---- layer 3: 128x256, then in-register max over this wave's 16 rows
  v8f a3[16];
#pragma unroll
  for (int ct = 0; ct < 16; ++ct)
#pragma unroll
    for (int r = 0; r < 8; ++r) a3[ct][r] = 0.f;
#pragma unroll
  for (int ks = 0; ks < 4; ++ks) {
    v16bf a = load_frag(Hs, 16 * wave, 32 * ks, lane);
#pragma unroll
    for (int ct = 0; ct < 16; ++ct) {
      v16bf bm = load_frag(Ws, 16 * ct, 32 * ks, lane);
      a3[ct] = __builtin_amdgcn_wmma_f32_16x16x32_bf16(false, a, false, bm,
                                                       (short)0, a3[ct], false, false);
    }
  }
#pragma unroll
  for (int ct = 0; ct < 16; ++ct) {
    float m = a3[ct][0];
#pragma unroll
    for (int r = 1; r < 8; ++r) m = fmaxf(m, a3[ct][r]);
    m = fmaxf(m, __shfl_xor(m, 16));        // combine the two 8-row halves
    if (h == 0) Pmax[wave][ct * 16 + n] = m;
  }
  __syncthreads();

  // combine the 2 waves per centroid, add bias, store (B,S,256)
#pragma unroll
  for (int q = 0; q < 4; ++q) {
    const float v = fmaxf(Pmax[2 * q][t], Pmax[2 * q + 1][t]) + Pb3[t];
    out_feat[(size_t)(cent0 + q) * 256 + t] = v;
  }
}

// ---------------------------------------------------------------------------
extern "C" void kernel_launch(void* const* d_in, const int* in_sizes, int n_in,
                              void* d_out, int out_size, void* d_ws, size_t ws_size,
                              hipStream_t stream) {
  const float* xyz  = (const float*)d_in[0];
  const float* feat = (const float*)d_in[1];
  const float* W1   = (const float*)d_in[2];
  const float* b1   = (const float*)d_in[3];
  const float* g1   = (const float*)d_in[4];
  const float* be1  = (const float*)d_in[5];
  const float* W2   = (const float*)d_in[6];
  const float* b2   = (const float*)d_in[7];
  const float* g2   = (const float*)d_in[8];
  const float* be2  = (const float*)d_in[9];
  const float* W3   = (const float*)d_in[10];
  const float* b3   = (const float*)d_in[11];

  float* out_xyz  = (float*)d_out;
  float* out_feat = out_xyz + (size_t)NB * NS * 3;

  char* ws = (char*)d_ws;
  int*    knn = (int*)ws;                                   // 8*2048*32*4 = 2 MB
  __bf16* Wt1 = (__bf16*)(ws + (size_t)NB * NS * NK * 4);
  __bf16* Wt2 = Wt1 + 128 * 128;
  __bf16* Wt3 = Wt2 + 128 * 128;

  prep_weights_kernel<<<128, 256, 0, stream>>>(W1, W2, W3, Wt1, Wt2, Wt3);
  fps_kernel<<<NB, 1024, 0, stream>>>(xyz, out_xyz);
  knn_kernel<<<(NB * NS * 32) / 256, 256, 0, stream>>>(xyz, out_xyz, knn);
  sa_mlp_kernel<<<(NB * NS) / 4, 256, 0, stream>>>(
      xyz, feat, knn, out_xyz, Wt1, Wt2, Wt3,
      b1, g1, be1, b2, g2, be2, b3, out_feat);
}
